// GINO_78692390797899
// MI455X (gfx1250) — matrix operand order repaired and checked by reading, not compiled, not run on hardware
//
#include <hip/hip_runtime.h>
#include <hip/hip_bf16.h>

// ---------------------------------------------------------------------------
// GINO forward for MI455X (gfx1250, wave32, WMMA).
// Encoder/decoder gather-MLPs (70% of FLOPs) on v_wmma_f32_16x16x32_f16 with
// LDS weights pre-swizzled into B-fragment order (contiguous v16h loads) and
// activations handed between layers in A-fragment order (contiguous v16h
// loads). FNO spectral conv = truncated DFT kernels, fp32 VALU with twiddle
// recurrences.
// ---------------------------------------------------------------------------

typedef __attribute__((ext_vector_type(16))) _Float16 v16h;
typedef __attribute__((ext_vector_type(8)))  _Float16 v8h;
typedef __attribute__((ext_vector_type(8)))  float    v8f;

#define PI_F 3.14159265358979f

// Branch-free GELU (exact-erf form via A&S 7.1.26, |err| ~ 1.5e-7).
__device__ __forceinline__ float gelu_f(float x) {
  float ax = fabsf(x) * 0.7071067811865475f;
  float t = __builtin_amdgcn_rcpf(1.0f + 0.3275911f * ax);
  float poly =
      ((((1.061405429f * t - 1.453152027f) * t + 1.421413741f) * t -
        0.284496736f) * t + 0.254829592f) * t;
  float erfa = 1.0f - poly * __expf(-ax * ax);
  float erfs = copysignf(erfa, x);
  return 0.5f * x * (1.0f + erfs);
}

__device__ __forceinline__ v8f wmma16x16x32(v16h a, v16h b, v8f c) {
  return __builtin_amdgcn_wmma_f32_16x16x32_f16(false, a, false, b, (short)0, c,
                                                false, false);
}

// ---------------------------------------------------------------------------
// Fragment layout helpers (ISA 7.12.2, 16-bit operands, wave32):
//  A 16x32: lane l row M=l&15; halves i=0..7 -> K=(l>=16?8:0)+i,
//           i=8..15 -> K=16+(l>=16?8:0)+(i-8).
//  B 32x16: lane l col N=l&15; halves i -> K same as A (table order).
//  C/D f32: vgpr j -> row M=j+(l>=16?8:0), col N=l&15.
// Weights are staged into LDS in B-fragment order:
//   sw[((kc*4+t)*32+l)*16+i] = W[(kc*32+(i&7)+(l>>4)*8+(i>>3)*16)*64 + (t*16+(l&15))]
// Activations are stored into LDS in A-fragment order:
//   hA[((m+16*selp)*2+kcp)*16+ip] where (kcp,selp,ip) decompose column n.
// ---------------------------------------------------------------------------
__device__ __forceinline__ int h_addr(int m, int n) {
  int kc = n >> 5, cc = n & 31;
  int sel = (cc >> 3) & 1;
  int i = (cc & 7) + ((cc >> 4) & 1) * 8;
  return ((m + 16 * sel) * 2 + kc) * 16 + i;
}

template <int KC>
__device__ __forceinline__ void stage_w(const float* __restrict__ W,
                                        _Float16* sw, int tid, int nthreads) {
  for (int d = tid; d < KC * 4 * 32 * 16; d += nthreads) {
    int i = d & 15, l = (d >> 4) & 31, t = (d >> 9) & 3, kc = d >> 11;
    int k = kc * 32 + (i & 7) + (l >> 4) * 8 + (i >> 3) * 16;
    int n = t * 16 + (l & 15);
    sw[d] = (_Float16)W[k * 64 + n];
  }
}

// Layer-1 A fragment: two aligned 16B runs from concat(rowA[0:96], rowB[0:96]).
__device__ __forceinline__ v16h load_a_l1(const _Float16* __restrict__ rowA,
                                          const _Float16* __restrict__ rowB,
                                          int kc, int sel) {
  int c0 = kc * 32 + sel * 8;
  int c1 = c0 + 16;
  v8h lo = *(const v8h*)((c0 < 96) ? (rowA + c0) : (rowB + (c0 - 96)));
  v8h hi = *(const v8h*)((c1 < 96) ? (rowA + c1) : (rowB + (c1 - 96)));
  return __builtin_shufflevector(lo, hi, 0, 1, 2, 3, 4, 5, 6, 7, 8, 9, 10, 11,
                                 12, 13, 14, 15);
}

// 64->64 WMMA layer: A from LDS fragment buffer, B from swizzled LDS weights.
__device__ __forceinline__ void layer64(const _Float16* hA, const _Float16* sw,
                                        int lane, v8f acc[4]) {
#pragma unroll
  for (int kc = 0; kc < 2; ++kc) {
    v16h a = *(const v16h*)(hA + (lane * 2 + kc) * 16);
#pragma unroll
    for (int t = 0; t < 4; ++t) {
      v16h b = *(const v16h*)(sw + ((kc * 4 + t) * 32 + lane) * 16);
      acc[t] = wmma16x16x32(a, b, acc[t]);
    }
  }
}

__device__ __forceinline__ void store_h(_Float16* hA, int ncol, int sel, int t,
                                        const v8f& acc, const float* B,
                                        bool dogelu) {
  int n = t * 16 + ncol;
  float bias = B[n];
#pragma unroll
  for (int j = 0; j < 8; ++j) {
    int m = j + sel * 8;
    float v = acc[j] + bias;
    if (dogelu) v = gelu_f(v);
    hA[h_addr(m, n)] = (_Float16)v;
  }
}

// ---------------------------------------------------------------------------
// Sinusoidal embedding: coords (n,3) -> f16 emb (n,96).
// ---------------------------------------------------------------------------
__global__ void embed_kernel(const float* __restrict__ coords,
                             _Float16* __restrict__ emb, int n) {
  int i = blockIdx.x * 256 + threadIdx.x;
  if (i >= n * 96) return;
  int ch = i % 96, pt = i / 96;
  int d = ch / 32, cc = ch % 32;
  int h = cc & 15;
  float freq = __expf(-(float)h * (9.210340371976184f / 16.0f)); // 1e4^(-h/16)
  float ang = coords[pt * 3 + d] * freq;
  float v = (cc >= 16) ? __cosf(ang) : __sinf(ang);
  emb[i] = (_Float16)v;
}

// ---------------------------------------------------------------------------
// Encoder GNO: one wave per latent point (8 waves/block, 4096 blocks).
// Fused MLP [192->64->64->64->3], masked mean pool over K=16 neighbors.
// ---------------------------------------------------------------------------
__global__ __launch_bounds__(256) void enc_gno_kernel(
    const _Float16* __restrict__ geomE, const _Float16* __restrict__ latE,
    const int* __restrict__ in_idx, const float* __restrict__ in_mask,
    const float* __restrict__ W1, const float* __restrict__ B1,
    const float* __restrict__ W2, const float* __restrict__ B2,
    const float* __restrict__ W3, const float* __restrict__ B3,
    const float* __restrict__ W4, const float* __restrict__ B4,
    float* __restrict__ in_p) {
  __shared__ _Float16 sW1[6 * 4 * 32 * 16];   // 192x64, B-frag order
  __shared__ _Float16 sW2[2 * 4 * 32 * 16];   // 64x64
  __shared__ _Float16 sW3[2 * 4 * 32 * 16];
  __shared__ _Float16 sW4[64 * 3];
  __shared__ float sB1[64], sB2[64], sB3[64], sB4v[3];
  __shared__ _Float16 hbuf[8][32 * 2 * 16];   // per-wave A-frag activations

  int tid = threadIdx.x;
  stage_w<6>(W1, sW1, tid, 256);
  stage_w<2>(W2, sW2, tid, 256);
  stage_w<2>(W3, sW3, tid, 256);
  for (int i = tid; i < 64 * 3; i += 256) sW4[i] = (_Float16)W4[i];
  if (tid < 64) { sB1[tid] = B1[tid]; sB2[tid] = B2[tid]; sB3[tid] = B3[tid]; }
  if (tid < 3) sB4v[tid] = B4[tid];
  __syncthreads();

  int wave = tid >> 5, lane = tid & 31;
  int sel = lane >> 4, mrow = lane & 15, ncol = lane & 15;
  int p = blockIdx.x * 8 + wave;
  int nb = in_idx[p * 16 + mrow];
  const _Float16* rowA = geomE + (size_t)nb * 96;
  const _Float16* rowB = latE + (size_t)p * 96;
  _Float16* hA = hbuf[wave];

  // ---- layer 1: 192 -> 64 ----
  v8f acc[4] = {};
#pragma unroll
  for (int kc = 0; kc < 6; ++kc) {
    v16h a = load_a_l1(rowA, rowB, kc, sel);
#pragma unroll
    for (int t = 0; t < 4; ++t) {
      v16h b = *(const v16h*)(sW1 + ((kc * 4 + t) * 32 + lane) * 16);
      acc[t] = wmma16x16x32(a, b, acc[t]);
    }
  }
#pragma unroll
  for (int t = 0; t < 4; ++t) store_h(hA, ncol, sel, t, acc[t], sB1, true);

  // ---- layer 2 ----
  v8f a2[4] = {};
  layer64(hA, sW2, lane, a2);
#pragma unroll
  for (int t = 0; t < 4; ++t) store_h(hA, ncol, sel, t, a2[t], sB2, true);

  // ---- layer 3 ----
  v8f a3[4] = {};
  layer64(hA, sW3, lane, a3);
#pragma unroll
  for (int t = 0; t < 4; ++t) store_h(hA, ncol, sel, t, a3[t], sB3, true);

  // ---- layer 4: 64 -> 3 (VALU), masked mean pool over 16 rows ----
  float o0 = 0.f, o1 = 0.f, o2 = 0.f;
  int kbase = sel * 32;
  for (int k = 0; k < 32; ++k) {
    int kk = kbase + k;
    float hv = (float)hA[h_addr(mrow, kk)];
    o0 += hv * (float)sW4[kk * 3 + 0];
    o1 += hv * (float)sW4[kk * 3 + 1];
    o2 += hv * (float)sW4[kk * 3 + 2];
  }
  o0 += __shfl_xor(o0, 16, 32);
  o1 += __shfl_xor(o1, 16, 32);
  o2 += __shfl_xor(o2, 16, 32);
  float msk = in_mask[p * 16 + mrow];
  o0 = (o0 + sB4v[0]) * msk;
  o1 = (o1 + sB4v[1]) * msk;
  o2 = (o2 + sB4v[2]) * msk;
  float cnt = msk;
  for (int off = 8; off > 0; off >>= 1) {
    o0 += __shfl_down(o0, off, 16);
    o1 += __shfl_down(o1, off, 16);
    o2 += __shfl_down(o2, off, 16);
    cnt += __shfl_down(cnt, off, 16);
  }
  if (lane == 0) {
    float c = fmaxf(cnt, 1.0f);
    in_p[p * 3 + 0] = o0 / c;
    in_p[p * 3 + 1] = o1 / c;
    in_p[p * 3 + 2] = o2 / c;
  }
}

// ---------------------------------------------------------------------------
// Lift MLP [3->128->64]; writes channel-major FNO field xr[c*32768+s].
// ---------------------------------------------------------------------------
__global__ __launch_bounds__(64) void lift_kernel(
    const float* __restrict__ in_p, const float* __restrict__ W1,
    const float* __restrict__ B1, const float* __restrict__ W2,
    const float* __restrict__ B2, float* __restrict__ xr) {
  __shared__ float h[128];
  int s = blockIdx.x, t = threadIdx.x;
  float c0 = in_p[s * 3 + 0], c1 = in_p[s * 3 + 1], c2 = in_p[s * 3 + 2];
  for (int j = t; j < 128; j += 64)
    h[j] = gelu_f(B1[j] + c0 * W1[j] + c1 * W1[128 + j] + c2 * W1[256 + j]);
  __syncthreads();
  float acc = B2[t];
  for (int j = 0; j < 128; ++j) acc += h[j] * W2[j * 64 + t];
  xr[(size_t)t * 32768 + s] = acc;
}

// ---------------------------------------------------------------------------
// FNO spectral conv as truncated DFTs (norm='forward' folded into z-DFT).
// Twiddles via complex rotation recurrence: 2 trans ops per thread total.
// ---------------------------------------------------------------------------
__global__ void fno_zdft(const float* __restrict__ xr, float2* __restrict__ Az) {
  int i = blockIdx.x * 256 + threadIdx.x;            // 64*32*32*17
  int kz = i % 17, r = i / 17;
  int iy = r & 31; r >>= 5;
  int ix = r & 31; int c = r >> 5;
  const float* src = xr + (size_t)c * 32768 + ix * 1024 + iy * 32;
  float th = -(PI_F / 16.0f) * (float)kz;
  float c0 = __cosf(th), s0 = __sinf(th);
  float cs = 1.f, sn = 0.f, sre = 0.f, sim = 0.f;
  for (int z = 0; z < 32; ++z) {
    float v = src[z];
    sre += v * cs;
    sim += v * sn;
    float nc = cs * c0 - sn * s0;
    sn = cs * s0 + sn * c0;
    cs = nc;
  }
  const float norm = 1.0f / 32768.0f;
  Az[i] = make_float2(sre * norm, sim * norm);
}

__global__ void fno_ydft(const float2* __restrict__ Az, float2* __restrict__ Ay) {
  int i = blockIdx.x * 256 + threadIdx.x;            // 64*32*16*17
  int kz = i % 17, r = i / 17;
  int j = r & 15; r >>= 4;
  int ix = r & 31; int c = r >> 5;
  int my = (j < 8) ? j : j + 16;
  float th = -(PI_F / 16.0f) * (float)my;
  float c0 = __cosf(th), s0 = __sinf(th);
  float cs = 1.f, sn = 0.f, sre = 0.f, sim = 0.f;
  for (int iy = 0; iy < 32; ++iy) {
    float2 v = Az[((c * 32 + ix) * 32 + iy) * 17 + kz];
    sre += v.x * cs - v.y * sn;
    sim += v.x * sn + v.y * cs;
    float nc = cs * c0 - sn * s0;
    sn = cs * s0 + sn * c0;
    cs = nc;
  }
  Ay[i] = make_float2(sre, sim);
}

__global__ void fno_xdft(const float2* __restrict__ Ay, float2* __restrict__ Ax) {
  int i = blockIdx.x * 256 + threadIdx.x;            // 64*16*16*17
  int kz = i % 17, r = i / 17;
  int jy = r & 15; r >>= 4;
  int jx = r & 15; int c = r >> 4;
  int mx = (jx < 8) ? jx : jx + 16;
  float th = -(PI_F / 16.0f) * (float)mx;
  float c0 = __cosf(th), s0 = __sinf(th);
  float cs = 1.f, sn = 0.f, sre = 0.f, sim = 0.f;
  for (int ix = 0; ix < 32; ++ix) {
    float2 v = Ay[((c * 32 + ix) * 16 + jy) * 17 + kz];
    sre += v.x * cs - v.y * sn;
    sim += v.x * sn + v.y * cs;
    float nc = cs * c0 - sn * s0;
    sn = cs * s0 + sn * c0;
    cs = nc;
  }
  Ax[i] = make_float2(sre, sim);
}

// Per-mode complex 64x64 matvec: one block per kept mode (4352 modes).
__global__ __launch_bounds__(64) void fno_modemix(
    const float2* __restrict__ Ax, const float* __restrict__ wr,
    const float* __restrict__ wi, float2* __restrict__ Bx) {
  __shared__ float2 xin[64];
  int mp = blockIdx.x;                               // 16*16*17
  int kz = mp % 17, jy = (mp / 17) & 15, jx = mp / (17 * 16);
  int o = threadIdx.x;
  xin[o] = Ax[((o * 16 + jx) * 16 + jy) * 17 + kz];
  __syncthreads();
  size_t wbase = (size_t)((jx * 16 + jy) * 17 + kz) * 4096;
  float are = 0.f, aim = 0.f;
  for (int i = 0; i < 64; ++i) {
    float wrv = wr[wbase + i * 64 + o], wiv = wi[wbase + i * 64 + o];
    float2 x = xin[i];
    are += x.x * wrv - x.y * wiv;
    aim += x.x * wiv + x.y * wrv;
  }
  Bx[((o * 16 + jx) * 16 + jy) * 17 + kz] = make_float2(are, aim);
}

__global__ void fno_xidft(const float2* __restrict__ Bx, float2* __restrict__ By) {
  int i = blockIdx.x * 256 + threadIdx.x;            // 64*32*16*17
  int kz = i % 17, r = i / 17;
  int jy = r & 15; r >>= 4;
  int x = r & 31; int c = r >> 5;
  float th = (PI_F / 16.0f) * (float)x;
  float c0 = __cosf(th), s0 = __sinf(th);
  // jx=0..7 -> mode jx (angle jx*th); jx=8..15 -> mode jx+16 (angle (jx+16)*th)
  float cs = 1.f, sn = 0.f, sre = 0.f, sim = 0.f;
  // jump factor for the +16 mode gap: e^{i*16*th}
  float cg = __cosf(16.0f * th), sg = __sinf(16.0f * th);
  for (int jx = 0; jx < 16; ++jx) {
    if (jx == 8) {  // advance phase by 16 steps
      float nc = cs * cg - sn * sg;
      sn = cs * sg + sn * cg;
      cs = nc;
    }
    float2 v = Bx[((c * 16 + jx) * 16 + jy) * 17 + kz];
    sre += v.x * cs - v.y * sn;
    sim += v.x * sn + v.y * cs;
    float nc = cs * c0 - sn * s0;
    sn = cs * s0 + sn * c0;
    cs = nc;
  }
  By[((c * 32 + x) * 16 + jy) * 17 + kz] = make_float2(sre, sim);
}

__global__ void fno_yidft(const float2* __restrict__ By, float2* __restrict__ Bz) {
  int i = blockIdx.x * 256 + threadIdx.x;            // 64*32*32*17
  int kz = i % 17, r = i / 17;
  int y = r & 31; r >>= 5;
  int x = r & 31; int c = r >> 5;
  float th = (PI_F / 16.0f) * (float)y;
  float c0 = __cosf(th), s0 = __sinf(th);
  float cg = __cosf(16.0f * th), sg = __sinf(16.0f * th);
  float cs = 1.f, sn = 0.f, sre = 0.f, sim = 0.f;
  for (int jy = 0; jy < 16; ++jy) {
    if (jy == 8) {
      float nc = cs * cg - sn * sg;
      sn = cs * sg + sn * cg;
      cs = nc;
    }
    float2 v = By[((c * 32 + x) * 16 + jy) * 17 + kz];
    sre += v.x * cs - v.y * sn;
    sim += v.x * sn + v.y * cs;
    float nc = cs * c0 - sn * s0;
    sn = cs * s0 + sn * c0;
    cs = nc;
  }
  Bz[i] = make_float2(sre, sim);
}

// irfft along z (Hermitian fold) + skip 1x1 conv + optional GELU.
__global__ void fno_final(const float2* __restrict__ Bz,
                          const float* __restrict__ xin,
                          const float* __restrict__ skip,
                          float* __restrict__ xout, int apply_gelu) {
  int i = blockIdx.x * 256 + threadIdx.x;            // 64*32768 = o*32768+s
  int s = i & 32767, o = i >> 15;
  int z = s & 31;
  int base = (o * 1024 + (s >> 5)) * 17;
  float th = (PI_F / 16.0f) * (float)z;
  float c0 = __cosf(th), s0 = __sinf(th);
  float cs = 1.f, sn = 0.f, acc = 0.f;
  for (int kz = 0; kz <= 16; ++kz) {
    float2 v = Bz[base + kz];
    float term = v.x * cs - v.y * sn;
    acc += (kz == 0 || kz == 16) ? term : 2.0f * term;
    float nc = cs * c0 - sn * s0;
    sn = cs * s0 + sn * c0;
    cs = nc;
  }
  for (int c = 0; c < 64; ++c)
    acc += xin[(size_t)c * 32768 + s] * skip[c * 64 + o];
  xout[i] = apply_gelu ? gelu_f(acc) : acc;
}

// ---------------------------------------------------------------------------
// Decoder GNO: one wave per output point (4 waves/block, 5000 blocks).
// Fused MLP [192->64->64->64->64], then agg = sum_r k2 * latent * mask.
// ---------------------------------------------------------------------------
__global__ __launch_bounds__(128) void dec_gno_kernel(
    const _Float16* __restrict__ latE, const _Float16* __restrict__ outqE,
    const int* __restrict__ out_idx, const float* __restrict__ out_mask,
    const float* __restrict__ W1, const float* __restrict__ B1,
    const float* __restrict__ W2, const float* __restrict__ B2,
    const float* __restrict__ W3, const float* __restrict__ B3,
    const float* __restrict__ W4, const float* __restrict__ B4,
    const float* __restrict__ latf, float* __restrict__ agg) {
  __shared__ _Float16 sW1[6 * 4 * 32 * 16];
  __shared__ _Float16 sW2[2 * 4 * 32 * 16];
  __shared__ _Float16 sW3[2 * 4 * 32 * 16];
  __shared__ _Float16 sW4[2 * 4 * 32 * 16];
  __shared__ float sB1[64], sB2[64], sB3[64], sB4[64];
  __shared__ _Float16 hbuf[4][32 * 2 * 16];

  int tid = threadIdx.x;
  stage_w<6>(W1, sW1, tid, 128);
  stage_w<2>(W2, sW2, tid, 128);
  stage_w<2>(W3, sW3, tid, 128);
  stage_w<2>(W4, sW4, tid, 128);
  if (tid < 64) {
    sB1[tid] = B1[tid]; sB2[tid] = B2[tid];
    sB3[tid] = B3[tid]; sB4[tid] = B4[tid];
  }
  __syncthreads();

  int wave = tid >> 5, lane = tid & 31;
  int sel = lane >> 4, mrow = lane & 15, ncol = lane & 15;
  int q = blockIdx.x * 4 + wave;
  bool q_ok = (q < 20000);
  int qc = q_ok ? q : 19999;                // clamp: keep EXEC uniform for WMMA
  int nb = out_idx[qc * 16 + mrow];
  const _Float16* rowA = latE + (size_t)nb * 96;
  const _Float16* rowB = outqE + (size_t)qc * 96;
  _Float16* hA = hbuf[wave];

  // ---- layer 1: 192 -> 64 ----
  v8f acc[4] = {};
#pragma unroll
  for (int kc = 0; kc < 6; ++kc) {
    v16h a = load_a_l1(rowA, rowB, kc, sel);
#pragma unroll
    for (int t = 0; t < 4; ++t) {
      v16h b = *(const v16h*)(sW1 + ((kc * 4 + t) * 32 + lane) * 16);
      acc[t] = wmma16x16x32(a, b, acc[t]);
    }
  }
#pragma unroll
  for (int t = 0; t < 4; ++t) store_h(hA, ncol, sel, t, acc[t], sB1, true);

  // ---- layers 2,3 ----
  v8f a2[4] = {};
  layer64(hA, sW2, lane, a2);
#pragma unroll
  for (int t = 0; t < 4; ++t) store_h(hA, ncol, sel, t, a2[t], sB2, true);
  v8f a3[4] = {};
  layer64(hA, sW3, lane, a3);
#pragma unroll
  for (int t = 0; t < 4; ++t) store_h(hA, ncol, sel, t, a3[t], sB3, true);

  // ---- layer 4 (linear) fused with masked aggregation over 16 rows ----
  v8f a4[4] = {};
  layer64(hA, sW4, lane, a4);
#pragma unroll
  for (int t = 0; t < 4; ++t) {
    int n = t * 16 + ncol;
    float part = 0.f;
#pragma unroll
    for (int j = 0; j < 8; ++j) {
      int r = j + sel * 8;
      int idx = out_idx[qc * 16 + r];
      float msk = out_mask[qc * 16 + r];
      float f = latf[(size_t)n * 32768 + idx];
      part += (a4[t][j] + sB4[n]) * msk * f;
    }
    part += __shfl_xor(part, 16, 32);
    if (q_ok && sel == 0) agg[(size_t)q * 64 + n] = part;
  }
}

// ---------------------------------------------------------------------------
// Projection MLP [64->256->1].
// ---------------------------------------------------------------------------
__global__ __launch_bounds__(64) void proj_kernel(
    const float* __restrict__ agg, const float* __restrict__ W1,
    const float* __restrict__ B1, const float* __restrict__ W2,
    const float* __restrict__ B2, float* __restrict__ out) {
  __shared__ float xs[64];
  __shared__ float partial[64];
  int q = blockIdx.x, t = threadIdx.x;
  xs[t] = agg[(size_t)q * 64 + t];
  __syncthreads();
  float p = 0.f;
  for (int jj = 0; jj < 4; ++jj) {
    int j = t * 4 + jj;
    float h = B1[j];
    for (int i = 0; i < 64; ++i) h += xs[i] * W1[i * 256 + j];
    p += gelu_f(h) * W2[j];
  }
  partial[t] = p;
  __syncthreads();
  for (int off = 32; off > 0; off >>= 1) {
    if (t < off) partial[t] += partial[t + off];
    __syncthreads();
  }
  if (t == 0) out[q] = partial[0] + B2[0];
}

// ---------------------------------------------------------------------------
// Host launcher. d_in order (dict / tree-flatten order of setup_inputs):
//  0 input_geom  1 latent_queries  2 output_queries  3 in_idx  4 in_mask
//  5 out_idx  6 out_mask  7..14 in_mlp(W,b x4)  15..22 out_mlp(W,b x4)
//  23..26 lift(W,b x2)  27..30 proj(W,b x2)  31 fno_wr  32 fno_wi  33 fno_skip
// ---------------------------------------------------------------------------
extern "C" void kernel_launch(void* const* d_in, const int* in_sizes, int n_in,
                              void* d_out, int out_size, void* d_ws,
                              size_t ws_size, hipStream_t stream) {
  const float* geom = (const float*)d_in[0];
  const float* latq = (const float*)d_in[1];
  const float* outq = (const float*)d_in[2];
  const int* in_idx = (const int*)d_in[3];
  const float* in_mask = (const float*)d_in[4];
  const int* out_idx = (const int*)d_in[5];
  const float* out_mask = (const float*)d_in[6];
  const float* iW[4] = {(const float*)d_in[7], (const float*)d_in[9],
                        (const float*)d_in[11], (const float*)d_in[13]};
  const float* iB[4] = {(const float*)d_in[8], (const float*)d_in[10],
                        (const float*)d_in[12], (const float*)d_in[14]};
  const float* oW[4] = {(const float*)d_in[15], (const float*)d_in[17],
                        (const float*)d_in[19], (const float*)d_in[21]};
  const float* oB[4] = {(const float*)d_in[16], (const float*)d_in[18],
                        (const float*)d_in[20], (const float*)d_in[22]};
  const float* lW1 = (const float*)d_in[23];
  const float* lB1 = (const float*)d_in[24];
  const float* lW2 = (const float*)d_in[25];
  const float* lB2 = (const float*)d_in[26];
  const float* pW1 = (const float*)d_in[27];
  const float* pB1 = (const float*)d_in[28];
  const float* pW2 = (const float*)d_in[29];
  const float* pB2 = (const float*)d_in[30];
  const float* fno_wr = (const float*)d_in[31];
  const float* fno_wi = (const float*)d_in[32];
  const float* fno_skip = (const float*)d_in[33];
  float* out = (float*)d_out;

  char* ws = (char*)d_ws;
  size_t off = 0;
  auto alloc = [&](size_t bytes) -> void* {
    void* p = (void*)(ws + off);
    off += (bytes + 255) & ~(size_t)255;
    return p;
  };
  _Float16* geomE = (_Float16*)alloc((size_t)30000 * 96 * 2);
  _Float16* latE = (_Float16*)alloc((size_t)32768 * 96 * 2);
  _Float16* outqE = (_Float16*)alloc((size_t)20000 * 96 * 2);
  float* in_p = (float*)alloc((size_t)32768 * 3 * 4);
  float* xrA = (float*)alloc((size_t)64 * 32768 * 4);
  float* xrB = (float*)alloc((size_t)64 * 32768 * 4);
  float2* Az = (float2*)alloc((size_t)64 * 32 * 32 * 17 * 8);
  float2* Ay = (float2*)alloc((size_t)64 * 32 * 16 * 17 * 8);
  float2* Ax = (float2*)alloc((size_t)64 * 16 * 16 * 17 * 8);
  float2* Bx = (float2*)alloc((size_t)64 * 16 * 16 * 17 * 8);
  float2* By = (float2*)alloc((size_t)64 * 32 * 16 * 17 * 8);
  float2* Bz = (float2*)alloc((size_t)64 * 32 * 32 * 17 * 8);
  float* agg = (float*)alloc((size_t)20000 * 64 * 4);

  embed_kernel<<<(32768 * 96 + 255) / 256, 256, 0, stream>>>(latq, latE, 32768);
  embed_kernel<<<(30000 * 96 + 255) / 256, 256, 0, stream>>>(geom, geomE, 30000);
  embed_kernel<<<(20000 * 96 + 255) / 256, 256, 0, stream>>>(outq, outqE, 20000);

  enc_gno_kernel<<<4096, 256, 0, stream>>>(geomE, latE, in_idx, in_mask,
                                           iW[0], iB[0], iW[1], iB[1], iW[2],
                                           iB[2], iW[3], iB[3], in_p);

  lift_kernel<<<32768, 64, 0, stream>>>(in_p, lW1, lB1, lW2, lB2, xrA);

  float* xin = xrA;
  float* xout = xrB;
  for (int l = 0; l < 4; ++l) {
    fno_zdft<<<(64 * 32 * 32 * 17) / 256, 256, 0, stream>>>(xin, Az);
    fno_ydft<<<(64 * 32 * 16 * 17) / 256, 256, 0, stream>>>(Az, Ay);
    fno_xdft<<<(64 * 16 * 16 * 17) / 256, 256, 0, stream>>>(Ay, Ax);
    fno_modemix<<<16 * 16 * 17, 64, 0, stream>>>(
        Ax, fno_wr + (size_t)l * 16 * 16 * 17 * 4096,
        fno_wi + (size_t)l * 16 * 16 * 17 * 4096, Bx);
    fno_xidft<<<(64 * 32 * 16 * 17) / 256, 256, 0, stream>>>(Bx, By);
    fno_yidft<<<(64 * 32 * 32 * 17) / 256, 256, 0, stream>>>(By, Bz);
    fno_final<<<(64 * 32768) / 256, 256, 0, stream>>>(
        Bz, xin, fno_skip + (size_t)l * 4096, xout, (l < 3) ? 1 : 0);
    float* tmp = xin; xin = xout; xout = tmp;
  }
  // After 4 swaps, result is back in xrA (== xin).

  dec_gno_kernel<<<5000, 128, 0, stream>>>(latE, outqE, out_idx, out_mask,
                                           oW[0], oB[0], oW[1], oB[1], oW[2],
                                           oB[2], oW[3], oB[3], xin, agg);

  proj_kernel<<<20000, 64, 0, stream>>>(agg, pW1, pB1, pW2, pB2, out);

  (void)in_sizes; (void)n_in; (void)out_size; (void)ws_size;
}
